// ModulatedConv2D_22651657519330
// MI455X (gfx1250) — matrix operand order repaired and checked by reading, not compiled, not run on hardware
//
#include <hip/hip_runtime.h>

typedef __attribute__((ext_vector_type(16))) _Float16 v16h;
typedef __attribute__((ext_vector_type(8)))  _Float16 v8h;
typedef __attribute__((ext_vector_type(8)))  float    v8f;

#define EPSV   1e-8f
#define NBATCH 16
#define CIN    512
#define COUT   512
#define HW     64
#define KTOT   (CIN * 9)       // 4608
#define BM     64              // Cout tile
#define BROWS  4               // image rows per block
#define BP     (BROWS * HW)    // 256 pixels per block
#define CCHUNK 32              // c per chunk (WMMA K)
#define NCHUNK (CIN / CCHUNK)  // 16
#define XROWS  (BROWS + 2)     // 6 (halo rows)
#define XCOLS  (HW + 2)        // 66 (halo cols)
#define XPOS   (XROWS * XCOLS) // 396
#define LDK    (CCHUNK + 8)    // 40 halfs: 80B stride, conflict-free & 16B-aligned

// ---------------------------------------------------------------------------
// d[n,o] = rsqrt( sum_c s[n,c]^2 * sum_r w[o,c,r]^2 + eps )
// ---------------------------------------------------------------------------
__global__ void demod_kernel(const float* __restrict__ s,
                             const float* __restrict__ w,
                             float* __restrict__ d)
{
    int idx = blockIdx.x * blockDim.x + threadIdx.x;   // n*COUT + o
    if (idx >= NBATCH * COUT) return;
    int n = idx >> 9;
    int o = idx & (COUT - 1);
    const float* wrow = w + (size_t)o * KTOT;
    const float* srow = s + n * CIN;
    float acc = 0.f;
    for (int c = 0; c < CIN; ++c) {
        float sv = srow[c];
        float wsum = 0.f;
        #pragma unroll
        for (int r = 0; r < 9; ++r) {
            float wv = wrow[c * 9 + r];
            wsum += wv * wv;
        }
        acc += sv * sv * wsum;
    }
    d[idx] = rsqrtf(acc + EPSV);
}

// ---------------------------------------------------------------------------
// Modulated conv as 9 shifted GEMMs over c-chunks of 32.
// Block: 256 threads (8 wave32). Tile: 64 Cout x 256 pixels (4 rows).
// Input slab Xs (modulated f16, with zero halo) staged once per c-chunk and
// reused by all 9 filter taps.
// ---------------------------------------------------------------------------
__global__ __launch_bounds__(256)
void modconv_kernel(const float* __restrict__ x,
                    const float* __restrict__ s,
                    const float* __restrict__ noise,
                    const float* __restrict__ w,
                    const float* __restrict__ bias,
                    const float* __restrict__ dcoef,
                    float* __restrict__ out)
{
    __shared__ _Float16 Xs[XPOS][LDK];      // 396*40*2  = 31.7 KB
    __shared__ _Float16 A2[2][BM][LDK];     // 2*64*40*2 = 10.2 KB
    __shared__ float    sSm[CIN];           //             2.0 KB

    const int tid   = threadIdx.x;
    const int n     = blockIdx.z;
    const int hbase = blockIdx.y * BROWS;
    const int obase = blockIdx.x * BM;

    const float* xbase = x + (size_t)n * CIN * HW * HW;

    // preload s[n, :]
    for (int i = tid; i < CIN; i += 256) sSm[i] = s[n * CIN + i];

    // A stager: thread -> o = tid>>2, 8 consecutive c at (tid&3)*8 (stride-9 gather in w)
    const int a_o = tid >> 2;
    const int a_c = (tid & 3) * 8;
    auto stageA = [&](int buf, int tap, int c0) {
        const float* src = w + (size_t)(obase + a_o) * KTOT + (size_t)(c0 + a_c) * 9 + tap;
        #pragma unroll
        for (int i = 0; i < 8; ++i)
            A2[buf][a_o][a_c + i] = (_Float16)src[(size_t)i * 9];
    };

    // X slab stager: one (row,col) position per iteration, all 32 c
    auto stageX = [&](int c0) {
        for (int pos = tid; pos < XPOS; pos += 256) {
            int row = pos / XCOLS;
            int col = pos - row * XCOLS;
            int hh  = hbase + row - 1;
            int ww  = col - 1;
            _Float16* dst = &Xs[pos][0];
            if ((unsigned)hh < HW && (unsigned)ww < HW) {
                const float* xp = xbase + (size_t)c0 * HW * HW + hh * HW + ww;
                #pragma unroll 8
                for (int c = 0; c < CCHUNK; ++c)
                    dst[c] = (_Float16)(sSm[c0 + c] * xp[(size_t)c * HW * HW]);
            } else {
                #pragma unroll 8
                for (int c = 0; c < CCHUNK; ++c) dst[c] = (_Float16)0.f;
            }
        }
    };

    // ---- wave -> output sub-tile mapping ----
    const int lane = tid & 31;
    const int wv   = tid >> 5;              // 0..7
    const int wv_o = (wv & 1) * 32;         // 0 / 32
    const int wv_r = wv >> 1;               // wave image row 0..3
    const int wv_p = wv_r * HW;             // 0 / 64 / 128 / 192
    const int m16  = lane & 15;
    const int hlf  = lane >> 4;

    // A fragment (16x32, M x K): lane m16; K halfs {hlf*8..+7} U {16+hlf*8..+7}
    auto ldA = [&](int buf, int orow) -> v16h {
        const _Float16* rp = &A2[buf][orow + m16][0];
        v8h lo = *(const v8h*)(rp + hlf * 8);
        v8h hi = *(const v8h*)(rp + 16 + hlf * 8);
        v16h r;
        #pragma unroll
        for (int i = 0; i < 8; ++i) { r[i] = lo[i]; r[8 + i] = hi[i]; }
        return r;
    };
    // B fragment (32x16, K x N): lane n = m16 selects slab position; K halfs {hlf*8..+7} U {16+hlf*8..+7}
    auto ldB = [&](int pos0) -> v16h {
        const _Float16* rp = &Xs[pos0 + m16][0] + hlf * 8;
        v8h lo = *(const v8h*)(rp);
        v8h hi = *(const v8h*)(rp + 16);
        v16h r;
        #pragma unroll
        for (int i = 0; i < 8; ++i) { r[i] = lo[i]; r[8 + i] = hi[i]; }
        return r;
    };

    v8f acc[2][4];
    #pragma unroll
    for (int oi = 0; oi < 2; ++oi)
        #pragma unroll
        for (int pi = 0; pi < 4; ++pi) acc[oi][pi] = (v8f){};

    __syncthreads();                         // sSm ready

    for (int cc = 0; cc < NCHUNK; ++cc) {
        const int c0 = cc * CCHUNK;
        stageX(c0);
        stageA(0, 0, c0);
        if (cc + 1 < NCHUNK)                 // warm next chunk's first w gather
            __builtin_prefetch(w + (size_t)(obase + a_o) * KTOT + (size_t)(c0 + CCHUNK + a_c) * 9, 0, 1);
        __syncthreads();                     // slab + tap-0 weights ready

        #pragma unroll
        for (int tap = 0; tap < 9; ++tap) {
            const int buf = tap & 1;
            if (tap + 1 < 9) stageA(buf ^ 1, tap + 1, c0);

            const int kho = tap / 3 - 1;
            const int kwo = tap - (tap / 3) * 3 - 1;
            // slab position of pixel column 0 for this wave row, shifted by tap
            const int bpos = (wv_r + 1 + kho) * XCOLS + (1 + kwo);

            v16h a0 = ldA(buf, wv_o);
            v16h a1 = ldA(buf, wv_o + 16);
            v16h b0 = ldB(bpos);
            v16h b1 = ldB(bpos + 16);
            v16h b2 = ldB(bpos + 32);
            v16h b3 = ldB(bpos + 48);

            acc[0][0] = __builtin_amdgcn_wmma_f32_16x16x32_f16(false, a0, false, b0, (short)0, acc[0][0], false, false);
            acc[0][1] = __builtin_amdgcn_wmma_f32_16x16x32_f16(false, a0, false, b1, (short)0, acc[0][1], false, false);
            acc[0][2] = __builtin_amdgcn_wmma_f32_16x16x32_f16(false, a0, false, b2, (short)0, acc[0][2], false, false);
            acc[0][3] = __builtin_amdgcn_wmma_f32_16x16x32_f16(false, a0, false, b3, (short)0, acc[0][3], false, false);
            acc[1][0] = __builtin_amdgcn_wmma_f32_16x16x32_f16(false, a1, false, b0, (short)0, acc[1][0], false, false);
            acc[1][1] = __builtin_amdgcn_wmma_f32_16x16x32_f16(false, a1, false, b1, (short)0, acc[1][1], false, false);
            acc[1][2] = __builtin_amdgcn_wmma_f32_16x16x32_f16(false, a1, false, b2, (short)0, acc[1][2], false, false);
            acc[1][3] = __builtin_amdgcn_wmma_f32_16x16x32_f16(false, a1, false, b3, (short)0, acc[1][3], false, false);

            __syncthreads();   // next-tap A staged; slab reads of this tap done
        }
        // tap-loop's final barrier also protects the next chunk's slab overwrite
    }

    // ---- fused epilogue: y = acc*d + b + noise, LeakyReLU(0.2) ----
    const float* drow = dcoef + n * COUT;
    #pragma unroll
    for (int oi = 0; oi < 2; ++oi) {
        #pragma unroll
        for (int pi = 0; pi < 4; ++pi) {
            #pragma unroll
            for (int r = 0; r < 8; ++r) {
                int o  = obase + wv_o + oi * 16 + r + 8 * hlf;  // C/D: M = r + 8*half
                int p  = wv_p + pi * 16 + m16;                  // N = lane % 16
                int hh = hbase + (p >> 6);
                int ww = p & 63;
                size_t oidx = (((size_t)n * COUT + o) * HW + hh) * HW + ww;
                float v = acc[oi][pi][r] * drow[o] + bias[o] + noise[oidx];
                out[oidx] = (v >= 0.f) ? v : 0.2f * v;
            }
        }
    }
}

// ---------------------------------------------------------------------------
extern "C" void kernel_launch(void* const* d_in, const int* in_sizes, int n_in,
                              void* d_out, int out_size, void* d_ws, size_t ws_size,
                              hipStream_t stream)
{
    (void)in_sizes; (void)n_in; (void)out_size; (void)ws_size;
    const float* x     = (const float*)d_in[0];
    const float* s     = (const float*)d_in[1];
    const float* noise = (const float*)d_in[2];
    const float* w     = (const float*)d_in[3];
    const float* b     = (const float*)d_in[4];
    float* out   = (float*)d_out;
    float* dcoef = (float*)d_ws;                 // 16*512 floats = 32 KB scratch

    demod_kernel<<<(NBATCH * COUT + 255) / 256, 256, 0, stream>>>(s, w, dcoef);

    dim3 grid(COUT / BM, HW / BROWS, NBATCH);    // 8 x 16 x 16 = 2048 blocks
    modconv_kernel<<<grid, 256, 0, stream>>>(x, s, noise, w, b, dcoef, out);
}
